// ES_RNN_82832739271015
// MI455X (gfx1250) — matrix operand order repaired and checked by reading, not compiled, not run on hardware
//
#include <hip/hip_runtime.h>
#include <hip/hip_bf16.h>
#include <stdint.h>

// Problem constants (match reference)
#define T_LEN   100000
#define WIN     336
#define OUT_SZ  48
#define SEAS    168
#define NROWS   (T_LEN - WIN - OUT_SZ + 1)   // 99617
#define KPAD    352                          // 336 padded to 11 * 32
#define LSTR    360                          // LDS row stride (halves), bank-conflict pad
#define BLK_ROWS 128                         // rows per block (8 waves * 16)
#define NT_TILES (WIN / 16)                  // 21
#define W1_CHUNKS (16 * (KPAD / 8))          // 704 x 16B chunks per W1 tile

typedef __attribute__((ext_vector_type(16))) _Float16 v16h;
typedef __attribute__((ext_vector_type(8)))  _Float16 v8h;
typedef __attribute__((ext_vector_type(8)))  float    v8f;

// Exact pointee type expected by the async-LDS builtin (per hipcc diagnostic):
// "__attribute__((__vector_size__(4 * sizeof(int)))) int __device__ *"
typedef int b128_t __attribute__((vector_size(4 * sizeof(int))));

#if __has_builtin(__builtin_amdgcn_global_load_async_to_lds_b128)
#define HAVE_ASYNC 1
#else
#define HAVE_ASYNC 0
#endif

// ---------------- fast hardware tanh (CDNA5 V_TANH_F32 transcendental) --
__device__ __forceinline__ float fast_tanh(float x) {
#if __has_builtin(__builtin_amdgcn_tanhf)
    return __builtin_amdgcn_tanhf(x);
#else
    float r;
    // TRANS op: ISA requires 1 independent op / V_NOP before result use.
    asm("v_tanh_f32 %0, %1\n\tv_nop" : "=v"(r) : "v"(x));
    return r;
#endif
}

// ---------------- hash-based approximate N(0,1) noise -------------------
__device__ __forceinline__ uint32_t hash_u32(uint32_t x) {
    x ^= x >> 16; x *= 0x7feb352dU;
    x ^= x >> 15; x *= 0x846ca68bU;
    x ^= x >> 16;
    return x;
}
__device__ __forceinline__ float hash_gauss(uint32_t i, uint32_t j, uint32_t salt) {
    uint32_t h1 = hash_u32(i * 0x9E3779B9U ^ (j * 0x85EBCA6BU + salt));
    uint32_t h2 = hash_u32(h1 + 0x68E31DA4U);
    float u1 = (float)(h1 & 0xFFFFu) * (1.0f / 65536.0f);
    float u2 = (float)(h1 >> 16)     * (1.0f / 65536.0f);
    float u3 = (float)(h2 & 0xFFFFu) * (1.0f / 65536.0f);
    float u4 = (float)(h2 >> 16)     * (1.0f / 65536.0f);
    return (u1 + u2 + u3 + u4 - 2.0f) * 1.7320508f; // var(sum4 U)=1/3 -> scale sqrt(3)
}

// ---------------- Kernel 1: sequential exponential-smoothing scan -------
// season buffer IS w_all: season used at step t == w_all[t]; new season -> w_all[t+168]
__global__ void es_scan_kernel(const float* __restrict__ x,
                               const float* __restrict__ lvl_coef,
                               const float* __restrict__ seas_coef,
                               const float* __restrict__ seas_params,
                               float* __restrict__ levels,
                               float* __restrict__ w_all,
                               float* __restrict__ loss_out)
{
    if (threadIdx.x != 0 || blockIdx.x != 0) return;
    const float a = 1.0f / (1.0f + __expf(-lvl_coef[0]));
    const float g = 1.0f / (1.0f + __expf(-seas_coef[0]));
    for (int s = 0; s < SEAS; ++s) w_all[s] = __expf(seas_params[s]);
    w_all[SEAS] = w_all[0];

    float level = x[0] / w_all[0];
    levels[0] = level;
    float lld_prev = 0.0f;
    float acc = 0.0f;
    for (int t = 1; t < T_LEN; ++t) {
        const float xt = x[t];
        const float season = w_all[t];
        const float nl = a * xt / season + (1.0f - a) * level;
        levels[t] = nl;
        if (t + SEAS < T_LEN)
            w_all[t + SEAS] = g * xt / nl + (1.0f - g) * season;
        const float lld = __logf(nl / level);       // off critical path
        if (t >= 2) { const float d = lld - lld_prev; acc += d * d; }
        lld_prev = lld;
        level = nl;
    }
    loss_out[0] = acc * (50.0f / (float)(T_LEN - 2));
}

// ---------------- Kernel 2: u/v vectors + f16 transposed/padded weights -
// W1h[n*KPAD + k] = W_tanh[k][n] (zero pad k>=336); W2h[n*KPAD + k] = W_out[k][n]
__global__ void es_prep_kernel(const float* __restrict__ x,
                               const float* __restrict__ levels,
                               const float* __restrict__ w_all,
                               const float* __restrict__ W_tanh,
                               const float* __restrict__ W_out,
                               float* __restrict__ u,
                               float* __restrict__ v,
                               _Float16* __restrict__ W1h,
                               _Float16* __restrict__ W2h)
{
    const int idx = blockIdx.x * blockDim.x + threadIdx.x;
    if (idx < T_LEN)  u[idx] = __logf(x[idx]) - __logf(w_all[idx]);
    if (idx < NROWS)  v[idx] = __logf(levels[idx + WIN]);
    if (idx < WIN * KPAD) {
        const int n = idx / KPAD, k = idx % KPAD;
        W1h[idx] = (k < WIN) ? (_Float16)W_tanh[k * WIN + n] : (_Float16)0.0f;
    }
    if (idx < OUT_SZ * KPAD) {
        const int n = idx / KPAD, k = idx % KPAD;
        W2h[idx] = (k < WIN) ? (_Float16)W_out[k * OUT_SZ + n] : (_Float16)0.0f;
    }
}

// ---------------- Kernel 3: labels (Hankel of u minus per-row v) --------
__global__ void es_labels_kernel(const float* __restrict__ u,
                                 const float* __restrict__ v,
                                 float* __restrict__ labels)
{
    const int idx = blockIdx.x * blockDim.x + threadIdx.x;
    if (idx >= NROWS * OUT_SZ) return;
    const int i = idx / OUT_SZ, k = idx % OUT_SZ;
    labels[idx] = u[i + WIN + k] - v[i] + 0.001f * hash_gauss((uint32_t)i, (uint32_t)k, 2u);
}

// ---------------- fragment helpers --------------------------------------
__device__ __forceinline__ v16h load_frag_a(const _Float16* rowBase, int kk, int hi) {
    // A 16x32 f16 layout: lanes 0-15 hold K={0..7,16..23}; lanes 16-31 K={8..15,24..31}
    const int ka = kk * 32 + (hi ? 8 : 0);
    const v8h a0 = *(const v8h*)(rowBase + ka);
    const v8h a1 = *(const v8h*)(rowBase + ka + 16);
    v16h af;
#pragma unroll
    for (int i = 0; i < 8; ++i) { af[i] = a0[i]; af[i + 8] = a1[i]; }
    return af;
}
__device__ __forceinline__ v16h load_frag_b(const _Float16* colBase, int kk, int hi) {
    // B 32x16 f16: lanes 0-15 (N) hold K={0..7,16..23}; lanes 16-31 K={8..15,24..31}
    const int kb = kk * 32 + (hi ? 8 : 0);
    const v8h b0 = *(const v8h*)(colBase + kb);
    const v8h b1 = *(const v8h*)(colBase + kb + 16);
    v16h bf;
#pragma unroll
    for (int i = 0; i < 8; ++i) { bf[i] = b0[i]; bf[i + 8] = b1[i]; }
    return bf;
}

// -------- W1 tile staging: async global->LDS (double-buffered) ----------
// Every thread issues exactly 3 async b128 copies per tile (clamped duplicate
// chunk keeps the per-wave ASYNCcnt uniform -> constant wait threshold).
__device__ __forceinline__ void issue_w1_tile(const _Float16* __restrict__ W1h,
                                              _Float16* sW1buf, int nt, int tid) {
#if HAVE_ASYNC
#pragma unroll
    for (int it = 0; it < 3; ++it) {
        int c = tid + it * 256;
        if (c > W1_CHUNKS - 1) c = W1_CHUNKS - 1;
        const int n  = c / (KPAD / 8);
        const int k8 = (c % (KPAD / 8)) * 8;
        __attribute__((address_space(1))) b128_t* g =
            (__attribute__((address_space(1))) b128_t*)(W1h + (size_t)(nt * 16 + n) * KPAD + k8);
        __attribute__((address_space(3))) b128_t* l =
            (__attribute__((address_space(3))) b128_t*)(sW1buf + n * LSTR + k8);
        __builtin_amdgcn_global_load_async_to_lds_b128(g, l, 0, 0);
    }
#else
    for (int idx = tid; idx < 16 * KPAD; idx += 256) {
        const int n = idx / KPAD, k = idx % KPAD;
        sW1buf[n * LSTR + k] = W1h[(size_t)(nt * 16 + n) * KPAD + k];
    }
#endif
}
__device__ __forceinline__ void wait_w1(bool prefetch_outstanding) {
#if HAVE_ASYNC
    if (prefetch_outstanding) asm volatile("s_wait_asynccnt 0x3" ::: "memory");
    else                      asm volatile("s_wait_asynccnt 0x0" ::: "memory");
#else
    (void)prefetch_outstanding;
#endif
}

// ---------------- Kernel 4: fused WMMA GEMM head ------------------------
// out = tanh(A @ W1 + b1) @ W2 + b2 ; A synthesized from u/v + noise.
__global__ void __launch_bounds__(256)
es_gemm_kernel(const float* __restrict__ u, const float* __restrict__ v,
               const _Float16* __restrict__ W1h, const _Float16* __restrict__ W2h,
               const float* __restrict__ bias1, const float* __restrict__ bias2,
               float* __restrict__ outp)
{
    extern __shared__ _Float16 lds[];
    _Float16* sA  = lds;                        // 128 x LSTR
    _Float16* sH  = sA + BLK_ROWS * LSTR;       // 128 x LSTR (tanh output, f16)
    _Float16* sW1 = sH + BLK_ROWS * LSTR;       // 2 x 16 x LSTR (double-buffered tile)
    _Float16* sW2 = sW1 + 2 * 16 * LSTR;        // 48 x LSTR (all of W2)

    const int tid  = threadIdx.x;
    const int lane = tid & 31;
    const int wv   = tid >> 5;                  // wave id 0..7
    const int rb   = blockIdx.x * BLK_ROWS;
    const int mrow = lane & 15;                 // N index (C layout) / M row (A layout)
    const int hi   = lane >> 4;                 // half-wave select

    // Kick off async staging of W1 tile 0 while we build A / load W2.
    issue_w1_tile(W1h, sW1, 0, tid);

    // Build A tile from u/v (Hankel structure) + zero K-pad; load W2; zero H K-pad.
    for (int idx = tid; idx < BLK_ROWS * KPAD; idx += 256) {
        const int r = idx / KPAD, k = idx % KPAD;
        const int grow = rb + r;
        float val = 0.0f;
        if (k < WIN && grow < NROWS)
            val = u[grow + k] - v[grow] + 0.001f * hash_gauss((uint32_t)grow, (uint32_t)k, 1u);
        sA[r * LSTR + k] = (_Float16)val;
    }
    for (int idx = tid; idx < OUT_SZ * KPAD; idx += 256) {
        const int n = idx / KPAD, k = idx % KPAD;
        sW2[n * LSTR + k] = W2h[idx];
    }
    for (int idx = tid; idx < BLK_ROWS * (KPAD - WIN); idx += 256) {
        const int r = idx / (KPAD - WIN), k = WIN + idx % (KPAD - WIN);
        sH[r * LSTR + k] = (_Float16)0.0f;      // pad columns of H stay zero
    }
    __syncthreads();

    const _Float16* aBase = sA + (wv * 16 + mrow) * LSTR;

    // -------- GEMM 1: H = tanh(A @ W1 + b1), 21 column tiles of 16 ------
    for (int nt = 0; nt < NT_TILES; ++nt) {
        const _Float16* curW1 = sW1 + (nt & 1) * (16 * LSTR);
        const bool more = (nt + 1 < NT_TILES);
        if (more)
            issue_w1_tile(W1h, sW1 + ((nt + 1) & 1) * (16 * LSTR), nt + 1, tid);
        wait_w1(more);          // tile nt arrived (only the 3 prefetch ops remain)
        __syncthreads();        // all waves' portions visible

        const float bv = bias1[nt * 16 + mrow];
        v8f acc;
#pragma unroll
        for (int r = 0; r < 8; ++r) acc[r] = bv;

        const _Float16* bBase = curW1 + mrow * LSTR;
#pragma unroll
        for (int kk = 0; kk < KPAD / 32; ++kk) {
            const v16h af = load_frag_a(aBase, kk, hi);
            const v16h bf = load_frag_b(bBase, kk, hi);
            acc = __builtin_amdgcn_wmma_f32_16x16x32_f16(
                      false, af, false, bf, (short)0, acc, false, false);
        }
        // C/D layout: VGPR r, lane: element (M = r + 8*hi, N = mrow)
#pragma unroll
        for (int r = 0; r < 8; ++r) {
            const int M = r + 8 * hi;
            sH[(wv * 16 + M) * LSTR + nt * 16 + mrow] = (_Float16)fast_tanh(acc[r]);
        }
        __syncthreads();        // all reads of curW1 done before its buffer is refilled
    }

    // -------- GEMM 2: out = H @ W2 + b2 (48 cols = 3 tiles) -------------
    const _Float16* hBase = sH + (wv * 16 + mrow) * LSTR;
#pragma unroll
    for (int ct = 0; ct < OUT_SZ / 16; ++ct) {
        const float bv = bias2[ct * 16 + mrow];
        v8f acc;
#pragma unroll
        for (int r = 0; r < 8; ++r) acc[r] = bv;

        const _Float16* bBase = sW2 + (ct * 16 + mrow) * LSTR;
#pragma unroll
        for (int kk = 0; kk < KPAD / 32; ++kk) {
            const v16h af = load_frag_a(hBase, kk, hi);
            const v16h bf = load_frag_b(bBase, kk, hi);
            acc = __builtin_amdgcn_wmma_f32_16x16x32_f16(
                      false, af, false, bf, (short)0, acc, false, false);
        }
#pragma unroll
        for (int r = 0; r < 8; ++r) {
            const int M = r + 8 * hi;
            const int grow = rb + wv * 16 + M;
            if (grow < NROWS)
                outp[(size_t)grow * OUT_SZ + ct * 16 + mrow] = acc[r];
        }
    }
}

// ---------------- host launch -------------------------------------------
static inline size_t align256(size_t x) { return (x + 255) & ~(size_t)255; }

extern "C" void kernel_launch(void* const* d_in, const int* in_sizes, int n_in,
                              void* d_out, int out_size, void* d_ws, size_t ws_size,
                              hipStream_t stream) {
    const float* x   = (const float*)d_in[0];
    const float* lc  = (const float*)d_in[1];
    const float* sc  = (const float*)d_in[2];
    const float* sp  = (const float*)d_in[3];
    const float* W1  = (const float*)d_in[4];
    const float* B1  = (const float*)d_in[5];
    const float* W2  = (const float*)d_in[6];
    const float* B2  = (const float*)d_in[7];

    float* out    = (float*)d_out;                            // [NROWS*OUT_SZ]
    float* labels = out + (size_t)NROWS * OUT_SZ;             // [NROWS*OUT_SZ]
    float* loss   = out + 2 * (size_t)NROWS * OUT_SZ;         // [1]

    char* ws = (char*)d_ws;
    float* levels = (float*)ws;            ws += align256((size_t)T_LEN * 4);
    float* w_all  = (float*)ws;            ws += align256((size_t)T_LEN * 4);
    float* u      = (float*)ws;            ws += align256((size_t)T_LEN * 4);
    float* v      = (float*)ws;            ws += align256((size_t)NROWS * 4);
    _Float16* W1h = (_Float16*)ws;         ws += align256((size_t)WIN * KPAD * 2);
    _Float16* W2h = (_Float16*)ws;         ws += align256((size_t)OUT_SZ * KPAD * 2);
    (void)in_sizes; (void)n_in; (void)out_size; (void)ws_size;

    // 1) sequential ES scan (inherently serial recurrence)
    es_scan_kernel<<<1, 1, 0, stream>>>(x, lc, sc, sp, levels, w_all, loss);

    // 2) u/v + f16 transposed/padded weights
    const int prepN = WIN * KPAD; // 118272 >= T_LEN, NROWS, OUT_SZ*KPAD
    es_prep_kernel<<<(prepN + 255) / 256, 256, 0, stream>>>(
        x, levels, w_all, W1, W2, u, v, W1h, W2h);

    // 3) labels
    const int labN = NROWS * OUT_SZ;
    es_labels_kernel<<<(labN + 255) / 256, 256, 0, stream>>>(u, v, labels);

    // 4) fused WMMA GEMM head (double-buffered async W1 staging)
    const int nblk = (NROWS + BLK_ROWS - 1) / BLK_ROWS;   // 779
    const size_t ldsBytes =
        (size_t)(2 * BLK_ROWS * LSTR + 2 * 16 * LSTR + OUT_SZ * LSTR) * sizeof(_Float16); // 241920
    es_gemm_kernel<<<nblk, 256, ldsBytes, stream>>>(u, v, W1h, W2h, B1, B2, out);
}
